// TransLayer_22488448762251
// MI455X (gfx1250) — compile-verified, matrix operand order checked by reading, not run on hardware
//
#include <hip/hip_runtime.h>
#include <hip/hip_bf16.h>

// ---------------------------------------------------------------------------
// Distance-biased MHA on MI455X (gfx1250, wave32, WMMA 16x16x32 f16->f32).
// N=6144, IN_DIM=256, DIM=128, HEADS=4, HD=32 (== one WMMA K-step).
// ---------------------------------------------------------------------------

typedef __attribute__((ext_vector_type(16))) _Float16 v16h;
typedef __attribute__((ext_vector_type(8)))  float    v8f;

#define NROWS  6144
#define INDIM  256
#define MDIM   128
#define NHEADS 4
#define HD     32

// Load a 16x32 f16 A-style fragment (also used for B = K^T and B = V via vT):
// lane l<16 : row l,  K = {0..7} at p, {16..23} at p+16
// lane l>=16: row l-16, K = {8..15},{24..31}  (caller bakes +8 into p)
__device__ __forceinline__ v16h ld_frag16(const _Float16* p) {
  union { v16h v; uint4 q[2]; } u;
  u.q[0] = *(const uint4*)(p);
  u.q[1] = *(const uint4*)(p + 16);
  return u.v;
}

__device__ __forceinline__ v8f wmma_f16(v16h a, v16h b, v8f c) {
  return __builtin_amdgcn_wmma_f32_16x16x32_f16(false, a, false, b, (short)0, c,
                                                false, false);
}

// ---------------------------------------------------------------------------
// Kernel 1: Q/K/V projection.  C[16x16] tile per wave, K-loop over 256.
// grid = (N/16, 3), block = (32, 8): wave w -> output cols [16w, 16w+16)
// mat 0 -> q [N,128] f16, mat 1 -> k [N,128] f16, mat 2 -> vT [128,N] f16
// ---------------------------------------------------------------------------
__global__ void qkv_proj_kernel(const float* __restrict__ x,
                                const float* __restrict__ Wq, const float* __restrict__ bq,
                                const float* __restrict__ Wk, const float* __restrict__ bk,
                                const float* __restrict__ Wv, const float* __restrict__ bv,
                                _Float16* __restrict__ qb, _Float16* __restrict__ kb,
                                _Float16* __restrict__ vTb) {
  const int mtile = blockIdx.x;
  const int mat   = blockIdx.y;          // 0=Q 1=K 2=V
  const int w     = threadIdx.y;         // col tile
  const int lane  = threadIdx.x;
  const int hlf   = lane >> 4;
  const int l     = lane & 15;
  const int row0  = mtile * 16;
  const int n0    = w * 16;

  const float* W = (mat == 0) ? Wq : (mat == 1) ? Wk : Wv;
  const float* b = (mat == 0) ? bq : (mat == 1) ? bk : bv;

  v8f c = {};
  for (int k0 = 0; k0 < INDIM; k0 += 32) {
    // A fragment from x (f32 -> f16)
    v16h a;
    {
      const float* p = x + (row0 + l) * INDIM + k0 + hlf * 8;
#pragma unroll
      for (int i = 0; i < 8; ++i) a[i] = (_Float16)p[i];
      p += 16;
#pragma unroll
      for (int i = 0; i < 8; ++i) a[8 + i] = (_Float16)p[i];
    }
    // B fragment from W (f32 -> f16), lane l holds output col n0+l
    v16h bf;
    {
      int kk = k0 + hlf * 8;
#pragma unroll
      for (int i = 0; i < 8; ++i) bf[i] = (_Float16)W[(kk + i) * MDIM + n0 + l];
      kk += 16;
#pragma unroll
      for (int i = 0; i < 8; ++i) bf[8 + i] = (_Float16)W[(kk + i) * MDIM + n0 + l];
    }
    c = wmma_f16(a, bf, c);
  }

  const float bias = b[n0 + l];
  if (mat < 2) {
    _Float16* dst = (mat == 0) ? qb : kb;
#pragma unroll
    for (int r = 0; r < 8; ++r) {
      const int row = row0 + r + hlf * 8;          // C layout: lanes>=16 hold M=r+8
      dst[row * MDIM + n0 + l] = (_Float16)(c[r] + bias);
    }
  } else {
#pragma unroll
    for (int r = 0; r < 8; ++r) {
      const int row = row0 + r + hlf * 8;
      vTb[(n0 + l) * NROWS + row] = (_Float16)(c[r] + bias);   // transposed V
    }
  }
}

// ---------------------------------------------------------------------------
// Kernel 2: flash attention with distance bias, 64-key chunks.
// grid = (N/64, HEADS), block = (32, 4).  Wave w owns query rows
// [64*bx + 16w, +16) of head blockIdx.y.
// Per chunk: 4 QK^T WMMAs -> online softmax (reductions amortized over 64
// keys) -> P staged in LDS f16 (C-layout -> A-layout) -> 4 PV WMMAs.
// ---------------------------------------------------------------------------
__global__ void attn_kernel(const _Float16* __restrict__ qb,
                            const _Float16* __restrict__ kb,
                            const _Float16* __restrict__ vTb,
                            const float* __restrict__ dist_w,
                            _Float16* __restrict__ attn) {
  const int h    = blockIdx.y;
  const int w    = threadIdx.y;
  const int lane = threadIdx.x;
  const int hlf  = lane >> 4;
  const int l    = lane & 15;
  const int q0   = blockIdx.x * 64 + w * 16;
  const float dws   = dist_w[h] * 0.01f;     // bias = dws * |i - j|
  const float scale = 0.1767766952966369f;   // 1/sqrt(32)

  // 16 rows x 64 cols per wave, stride 72 keeps b128 fragment loads 16B-aligned
  __shared__ __align__(16) _Float16 Plds[4][16][72];
  _Float16* myP = &Plds[w][0][0];

  // Persistent Q fragment (A operand), head slice h*32 .. h*32+31
  const _Float16* qp = qb + (q0 + l) * MDIM + h * HD + hlf * 8;
  const v16h qa = ld_frag16(qp);

  float m_run[8], l_run[8];
#pragma unroll
  for (int r = 0; r < 8; ++r) { m_run[r] = -1e30f; l_run[r] = 0.0f; }
  v8f o0 = {}, o1 = {};

  for (int j0 = 0; j0 < NROWS; j0 += 64) {
    // --- S = Q K^T over 4 key tiles of 16 ---
    v8f s[4];
#pragma unroll
    for (int t = 0; t < 4; ++t) {
      const _Float16* kp = kb + (j0 + 16 * t + l) * MDIM + h * HD + hlf * 8;
      const v16h kf = ld_frag16(kp);
      v8f z = {};
      s[t] = wmma_f16(qa, kf, z);     // keys j0+16t .. j0+16t+15
    }

    // --- scale + distance bias; online softmax ---
#pragma unroll
    for (int r = 0; r < 8; ++r) {
      const float qi = (float)(q0 + r + hlf * 8);
#pragma unroll
      for (int t = 0; t < 4; ++t) {
        const float kj = (float)(j0 + 16 * t + l);
        s[t][r] = s[t][r] * scale - dws * fabsf(qi - kj);
      }
      float m = fmaxf(fmaxf(s[0][r], s[1][r]), fmaxf(s[2][r], s[3][r]));
#pragma unroll
      for (int mask = 1; mask <= 8; mask <<= 1)
        m = fmaxf(m, __shfl_xor(m, mask, 32));     // stays within 16-lane half
      const float mnew  = fmaxf(m_run[r], m);
      const float alpha = __expf(m_run[r] - mnew);
      m_run[r] = mnew;

      float p[4], rs = 0.0f;
#pragma unroll
      for (int t = 0; t < 4; ++t) {
        p[t] = __expf(s[t][r] - mnew);
        rs += p[t];
      }
#pragma unroll
      for (int mask = 1; mask <= 8; mask <<= 1)
        rs += __shfl_xor(rs, mask, 32);
      l_run[r] = l_run[r] * alpha + rs;
      o0[r] *= alpha;
      o1[r] *= alpha;

      const int prow = (r + hlf * 8) * 72;
#pragma unroll
      for (int t = 0; t < 4; ++t)
        myP[prow + 16 * t + l] = (_Float16)p[t];
    }
    __asm__ volatile("" ::: "memory");   // keep DS stores before DS loads (in-order per wave)

    // --- O += P V : two K-steps of 32 keys each ---
#pragma unroll
    for (int ks = 0; ks < 2; ++ks) {
      const v16h pa  = ld_frag16(myP + l * 72 + ks * 32 + hlf * 8);
      const v16h vf0 = ld_frag16(vTb + (h * HD + l)      * NROWS + j0 + ks * 32 + hlf * 8);
      const v16h vf1 = ld_frag16(vTb + (h * HD + 16 + l) * NROWS + j0 + ks * 32 + hlf * 8);
      o0 = wmma_f16(pa, vf0, o0);   // head dims 0..15
      o1 = wmma_f16(pa, vf1, o1);   // head dims 16..31
    }
    __asm__ volatile("" ::: "memory");   // next iter's P stores stay after these loads
  }

  // --- normalize and emit f16 attention output [N, 128] ---
#pragma unroll
  for (int r = 0; r < 8; ++r) {
    const int row = q0 + r + hlf * 8;
    const float inv = 1.0f / l_run[r];
    attn[row * MDIM + h * HD + l]      = (_Float16)(o0[r] * inv);
    attn[row * MDIM + h * HD + 16 + l] = (_Float16)(o1[r] * inv);
  }
}

// ---------------------------------------------------------------------------
// Kernel 3: output projection  out = attn @ Wo + bo  (f32 out).
// grid = (N/16), block = (32, 8): wave w -> cols [16w, 16w+16).
// ---------------------------------------------------------------------------
__global__ void out_proj_kernel(const _Float16* __restrict__ attn,
                                const float* __restrict__ Wo,
                                const float* __restrict__ bo,
                                float* __restrict__ out) {
  const int mtile = blockIdx.x;
  const int w     = threadIdx.y;
  const int lane  = threadIdx.x;
  const int hlf   = lane >> 4;
  const int l     = lane & 15;
  const int row0  = mtile * 16;
  const int n0    = w * 16;

  v8f c = {};
  for (int k0 = 0; k0 < MDIM; k0 += 32) {
    const v16h a = ld_frag16(attn + (row0 + l) * MDIM + k0 + hlf * 8);
    v16h bf;
    {
      int kk = k0 + hlf * 8;
#pragma unroll
      for (int i = 0; i < 8; ++i) bf[i] = (_Float16)Wo[(kk + i) * MDIM + n0 + l];
      kk += 16;
#pragma unroll
      for (int i = 0; i < 8; ++i) bf[8 + i] = (_Float16)Wo[(kk + i) * MDIM + n0 + l];
    }
    c = wmma_f16(a, bf, c);
  }

  const float bias = bo[n0 + l];
#pragma unroll
  for (int r = 0; r < 8; ++r) {
    const int row = row0 + r + hlf * 8;
    out[row * MDIM + n0 + l] = c[r] + bias;
  }
}

// ---------------------------------------------------------------------------
extern "C" void kernel_launch(void* const* d_in, const int* in_sizes, int n_in,
                              void* d_out, int out_size, void* d_ws, size_t ws_size,
                              hipStream_t stream) {
  (void)in_sizes; (void)n_in; (void)out_size; (void)ws_size;

  const float* x   = (const float*)d_in[0];
  const float* Wq  = (const float*)d_in[1];
  const float* bq  = (const float*)d_in[2];
  const float* Wk  = (const float*)d_in[3];
  const float* bk  = (const float*)d_in[4];
  const float* Wv  = (const float*)d_in[5];
  const float* bv  = (const float*)d_in[6];
  const float* Wo  = (const float*)d_in[7];
  const float* bo  = (const float*)d_in[8];
  const float* dwv = (const float*)d_in[9];
  float* out = (float*)d_out;

  char* ws = (char*)d_ws;
  const size_t MAT = (size_t)NROWS * MDIM * sizeof(_Float16);   // 1.5 MB
  _Float16* qb   = (_Float16*)(ws + 0 * MAT);
  _Float16* kb   = (_Float16*)(ws + 1 * MAT);
  _Float16* vTb  = (_Float16*)(ws + 2 * MAT);
  _Float16* atnb = (_Float16*)(ws + 3 * MAT);

  qkv_proj_kernel<<<dim3(NROWS / 16, 3), dim3(32, 8), 0, stream>>>(
      x, Wq, bq, Wk, bk, Wv, bv, qb, kb, vTb);

  attn_kernel<<<dim3(NROWS / 64, NHEADS), dim3(32, 4), 0, stream>>>(
      qb, kb, vTb, dwv, atnb);

  out_proj_kernel<<<dim3(NROWS / 16), dim3(32, 8), 0, stream>>>(
      atnb, Wo, bo, out);
}
